// MLAAttention_82360292868232
// MI455X (gfx1250) — compile-verified, matrix operand order checked by reading, not run on hardware
//
#include <hip/hip_runtime.h>
#include <hip/hip_bf16.h>
#include <math.h>

// ---------------- problem constants ----------------
#define T_SEQ  2048
#define H_DIM  4096
#define NHEAD  32
#define QLORA  1536
#define RDIM   512
#define DKN    128
#define DKR    64
#define QKH    192          // DKN + DKR
#define DQK    576          // RDIM + DKR (combined latent+rope attention dim)
#define SCALE_ATTN 0.07216878364870323f   // 192^-0.5
#define EPSR   1e-6f

#define APITCH 40           // LDS pitch (u16) for [rows][32] tiles: 80B rows, 16B aligned,
                            // r*20 dwords mod 64 banks -> 16 distinct banks, conflict-free
#define BPITCH 136          // LDS pitch for row-major [32][128] B tiles (272B rows)

typedef unsigned short u16;
typedef __attribute__((ext_vector_type(16))) __bf16 v16bf;
typedef __attribute__((ext_vector_type(8)))  float  v8f;

union B16Frag { u16 u[16]; v16bf v; };

__device__ __forceinline__ u16 f2b(float x) {
  unsigned u = __float_as_uint(x);
  unsigned r = u + 0x7FFFu + ((u >> 16) & 1u);   // round-to-nearest-even
  return (u16)(r >> 16);
}

// ---- CDNA5 async memory->LDS path (ASYNCcnt-tracked, no VGPR round trip) ----
__device__ __forceinline__ unsigned lds_off(const void* p) {
  return (unsigned)(unsigned long long)(__attribute__((address_space(3))) const char*)p;
}
__device__ __forceinline__ void async_ld_b128(unsigned ldsOff, const void* g) {
  asm volatile("global_load_async_to_lds_b128 %0, %1, off" :: "v"(ldsOff), "v"(g) : "memory");
}
__device__ __forceinline__ void wait_async0() {
  asm volatile("s_wait_asynccnt 0x0" ::: "memory");
}

// A-matrix frag (16x32 bf16, ISA 7.12.2): lane&15 = row, lane>>4 selects K half,
// VGPR v holds K pair kb,kb+1 with kb = 2v + (v>=4 ? 8 : 0) + 8*half.
__device__ __forceinline__ v16bf load_frag_a(const u16* t, int ld) {
  const int lane = threadIdx.x & 31;
  const int half = lane >> 4;
  const int r    = lane & 15;
  B16Frag f;
#pragma unroll
  for (int v = 0; v < 8; ++v) {
    int kb = 2 * v + ((v >= 4) ? 8 : 0) + 8 * half;
    f.u[2 * v]     = t[r * ld + kb];
    f.u[2 * v + 1] = t[r * ld + kb + 1];
  }
  return f.v;
}

// B-matrix frag from row-major [K][N] storage (lane&15 = column).
__device__ __forceinline__ v16bf load_frag_b_rm(const u16* t, int ld) {
  const int lane = threadIdx.x & 31;
  const int half = lane >> 4;
  const int c    = lane & 15;
  B16Frag f;
#pragma unroll
  for (int v = 0; v < 8; ++v) {
    int kb = 2 * v + ((v >= 4) ? 8 : 0) + 8 * half;
    f.u[2 * v]     = t[kb * ld + c];
    f.u[2 * v + 1] = t[(kb + 1) * ld + c];
  }
  return f.v;
}

#define WMMA_BF16(a, b, c) \
  __builtin_amdgcn_wmma_f32_16x16x32_bf16(false, (a), false, (b), (short)0, (c), false, false)

// ---------------- cast fp32 -> bf16 (8 elems / thread / iter, b128 in/out) ----------------
__global__ __launch_bounds__(256) void cast_f32_bf16(const float* __restrict__ in,
                                                     u16* __restrict__ out, long long n8) {
  long long i = (long long)blockIdx.x * blockDim.x + threadIdx.x;
  long long stride = (long long)gridDim.x * blockDim.x;
  for (; i < n8; i += stride) {
    const float4 a = ((const float4*)in)[2 * i];
    const float4 b = ((const float4*)in)[2 * i + 1];
    ushort4 lo, hi;
    lo.x = f2b(a.x); lo.y = f2b(a.y); lo.z = f2b(a.z); lo.w = f2b(a.w);
    hi.x = f2b(b.x); hi.y = f2b(b.y); hi.z = f2b(b.z); hi.w = f2b(b.w);
    ((ushort4*)out)[2 * i]     = lo;
    ((ushort4*)out)[2 * i + 1] = hi;
  }
}

// ---------------- generic bf16 GEMM, fp32 accumulate, async double-buffered ----------------
// C[M x NN] = A[M x K] * B.  TRANSB: B stored as [NN x K] row-major.
// 256 threads = 8 waves; 128x128 C tile; wave = 64x32 stripe = 4x2 WMMA subtiles.
template <bool TRANSB, bool OUTBF>
__global__ __launch_bounds__(256) void gemm_bf16(
    const u16* __restrict__ A, const u16* __restrict__ B, void* __restrict__ Cp,
    int M, int NN, int K, int lda, int ldb, int ldc,
    long long aBatch, long long bBatch, long long cBatch) {
  const int z = blockIdx.z;
  A += (long long)z * aBatch;
  B += (long long)z * bBatch;
  const int m0 = blockIdx.y * 128, n0 = blockIdx.x * 128;
  __shared__ __align__(16) u16 As[2][128 * APITCH];
  __shared__ __align__(16) u16 Bs[2][128 * APITCH];  // TRANSB: [col][APITCH]; else [k][BPITCH]
  const int tid  = threadIdx.x;
  const int wid  = tid >> 5, lane = tid & 31;
  const int half = lane >> 4, cl = lane & 15;
  const int wm = wid >> 2, wn = wid & 3;

  // A chunk map: 512 chunks of 8 elems; thread -> chunks {tid, tid+256}
  const int ar0 = tid >> 2,         ac0 = (tid & 3) * 8;
  const int ar1 = (tid + 256) >> 2, ac1 = ((tid + 256) & 3) * 8;
  // B chunk maps
  const int bk0 = tid >> 4,         bc0 = (tid & 15) * 8;           // row-major B
  const int bk1 = (tid + 256) >> 4, bc1 = ((tid + 256) & 15) * 8;
  // clamped global indices (OOB -> 0; garbage only reaches unstored C rows/cols)
  const int gmA0 = (m0 + ar0 < M) ? (m0 + ar0) : 0;
  const int gmA1 = (m0 + ar1 < M) ? (m0 + ar1) : 0;
  const int gnT0 = (n0 + ar0 < NN) ? (n0 + ar0) : 0;   // TRANSB row index
  const int gnT1 = (n0 + ar1 < NN) ? (n0 + ar1) : 0;
  const int gnR0 = (n0 + bc0 < NN) ? (n0 + bc0) : 0;   // row-major col index
  const int gnR1 = (n0 + bc1 < NN) ? (n0 + bc1) : 0;

  auto issue_tile = [&](int buf, int k0) {
    async_ld_b128(lds_off(&As[buf][ar0 * APITCH + ac0]), &A[(size_t)gmA0 * lda + k0 + ac0]);
    async_ld_b128(lds_off(&As[buf][ar1 * APITCH + ac1]), &A[(size_t)gmA1 * lda + k0 + ac1]);
    if (TRANSB) {
      async_ld_b128(lds_off(&Bs[buf][ar0 * APITCH + ac0]), &B[(size_t)gnT0 * ldb + k0 + ac0]);
      async_ld_b128(lds_off(&Bs[buf][ar1 * APITCH + ac1]), &B[(size_t)gnT1 * ldb + k0 + ac1]);
    } else {
      async_ld_b128(lds_off(&Bs[buf][bk0 * BPITCH + bc0]), &B[(size_t)(k0 + bk0) * ldb + gnR0]);
      async_ld_b128(lds_off(&Bs[buf][bk1 * BPITCH + bc1]), &B[(size_t)(k0 + bk1) * ldb + gnR1]);
    }
  };

  v8f acc[4][2];
  v8f vzero = {};
#pragma unroll
  for (int i = 0; i < 4; ++i)
#pragma unroll
    for (int j = 0; j < 2; ++j) acc[i][j] = vzero;

  issue_tile(0, 0);
  wait_async0();
  __syncthreads();
  int cur = 0;

  for (int k0 = 32; k0 <= K; k0 += 32) {
    const bool more = (k0 < K);
    if (more) issue_tile(cur ^ 1, k0);   // async prefetch next tile into other buffer
    // ---- compute current tile ----
    v16bf af[4], bf2[2];
#pragma unroll
    for (int sm = 0; sm < 4; ++sm)
      af[sm] = load_frag_a(&As[cur][(wm * 64 + sm * 16) * APITCH], APITCH);
#pragma unroll
    for (int sn = 0; sn < 2; ++sn) {
      if (TRANSB) bf2[sn] = load_frag_a(&Bs[cur][(wn * 32 + sn * 16) * APITCH], APITCH);
      else        bf2[sn] = load_frag_b_rm(&Bs[cur][wn * 32 + sn * 16], BPITCH);
    }
#pragma unroll
    for (int sm = 0; sm < 4; ++sm)
#pragma unroll
      for (int sn = 0; sn < 2; ++sn) acc[sm][sn] = WMMA_BF16(af[sm], bf2[sn], acc[sm][sn]);
    if (more) wait_async0();
    __syncthreads();
    cur ^= 1;
  }

#pragma unroll
  for (int sm = 0; sm < 4; ++sm)
#pragma unroll
    for (int sn = 0; sn < 2; ++sn)
#pragma unroll
      for (int i = 0; i < 8; ++i) {
        int m  = m0 + wm * 64 + sm * 16 + half * 8 + i;
        int nn = n0 + wn * 32 + sn * 16 + cl;
        if (m < M && nn < NN) {
          if (OUTBF)
            ((u16*)Cp)[(size_t)z * cBatch + (size_t)m * ldc + nn] = f2b(acc[sm][sn][i]);
          else
            ((float*)Cp)[(size_t)z * cBatch + (size_t)m * ldc + nn] = acc[sm][sn][i];
        }
      }
}

// ---------------- RMSNorm (per row), bf16 out (+ optional fp32 out) ----------------
__global__ __launch_bounds__(256) void rmsnorm_kernel(
    const float* __restrict__ in, int inStride, int D, const float* __restrict__ w,
    u16* __restrict__ outB, int outBStride, float* __restrict__ outF, int outFStride) {
  const int t = blockIdx.x;
  const float* row = in + (size_t)t * inStride;
  __shared__ float red[8];
  const int tid = threadIdx.x;
  float s = 0.f;
  for (int i = tid * 4; i < D; i += 1024) {
    float4 v = *(const float4*)&row[i];
    s += v.x * v.x + v.y * v.y + v.z * v.z + v.w * v.w;
  }
  for (int o = 16; o > 0; o >>= 1) s += __shfl_xor(s, o);
  if ((tid & 31) == 0) red[tid >> 5] = s;
  __syncthreads();
  float tot = 0.f;
#pragma unroll
  for (int k = 0; k < 8; ++k) tot += red[k];
  float rstd = rsqrtf(tot / (float)D + EPSR);
  for (int i = tid * 4; i < D; i += 1024) {
    float4 v = *(const float4*)&row[i];
    float4 g = *(const float4*)&w[i];
    float4 o;
    o.x = v.x * rstd * g.x; o.y = v.y * rstd * g.y;
    o.z = v.z * rstd * g.z; o.w = v.w * rstd * g.w;
    ushort4 ob; ob.x = f2b(o.x); ob.y = f2b(o.y); ob.z = f2b(o.z); ob.w = f2b(o.w);
    *(ushort4*)&outB[(size_t)t * outBStride + i] = ob;
    if (outF) *(float4*)&outF[(size_t)t * outFStride + i] = o;
  }
}

// ---------------- q post: split nope, interleaved RoPE on q_pe ----------------
__global__ void q_post(const float* __restrict__ qfull, u16* __restrict__ qnope,
                       u16* __restrict__ combq) {
  const int t = blockIdx.x, n = blockIdx.y, tid = threadIdx.x;  // block 64
  const float* src = qfull + ((size_t)t * NHEAD + n) * QKH;
  if (tid < 32) {
    float4 v = *(const float4*)&src[tid * 4];
    ushort4 ob; ob.x = f2b(v.x); ob.y = f2b(v.y); ob.z = f2b(v.z); ob.w = f2b(v.w);
    *(ushort4*)&qnope[((size_t)t * NHEAD + n) * DKN + tid * 4] = ob;
  } else {
    int i = tid - 32;   // rope pair index 0..31
    float inv = powf(10000.0f, -((float)(2 * i)) / (float)DKR);
    float ang = (float)t * inv;
    float c = cosf(ang), sn = sinf(ang);
    float x1 = src[DKN + 2 * i], x2 = src[DKN + 2 * i + 1];
    u16* dst = combq + ((size_t)t * NHEAD + n) * DQK + RDIM;
    dst[2 * i]     = f2b(x1 * c - x2 * sn);
    dst[2 * i + 1] = f2b(x1 * sn + x2 * c);
  }
}

// ---------------- k_pe RoPE -> kv_bf tail + kv_fused fp32 tail ----------------
__global__ void kpe_rope(const float* __restrict__ kva, u16* __restrict__ kvbf,
                         float* __restrict__ kvfused) {
  const int t = blockIdx.x, i = threadIdx.x;   // block 32 (pairs)
  const float* src = kva + (size_t)t * DQK + RDIM;
  float inv = powf(10000.0f, -((float)(2 * i)) / (float)DKR);
  float ang = (float)t * inv;
  float c = cosf(ang), sn = sinf(ang);
  float x1 = src[2 * i], x2 = src[2 * i + 1];
  float o1 = x1 * c - x2 * sn, o2 = x1 * sn + x2 * c;
  kvbf[(size_t)t * DQK + RDIM + 2 * i]     = f2b(o1);
  kvbf[(size_t)t * DQK + RDIM + 2 * i + 1] = f2b(o2);
  kvfused[(size_t)t * DQK + RDIM + 2 * i]     = o1;
  kvfused[(size_t)t * DQK + RDIM + 2 * i + 1] = o2;
}

// ---------------- attention pass 1: per-(t,n) softmax max m and sum l ----------------
__global__ __launch_bounds__(256) void attn_pass1(
    const u16* __restrict__ qc, const u16* __restrict__ kv,
    float* __restrict__ mOut, float* __restrict__ lOut) {
  const int qb = blockIdx.x, n = blockIdx.y;
  const int q0 = qb * 64;
  __shared__ __align__(16) u16 Qs[2][64 * APITCH];
  __shared__ __align__(16) u16 Ks[2][64 * APITCH];
  __shared__ float mrun[64], lrun[64], mnew[64];
  __shared__ float pmax[2][64], psum[2][64];
  const int tid = threadIdx.x;
  const int wid = tid >> 5, lane = tid & 31;
  const int half = lane >> 4, cl = lane & 15;
  const int wm = wid >> 1, wn = wid & 1;   // 4 row groups x 2 col groups
  const int sr = tid >> 2, sc = (tid & 3) * 8;  // staging chunk: 1 per thread per tile
  if (tid < 64) { mrun[tid] = -__builtin_inff(); lrun[tid] = 0.f; }
  __syncthreads();
  const size_t qrow = ((size_t)(q0 + sr) * NHEAD + n) * DQK + sc;
  const unsigned qdst0 = lds_off(&Qs[0][sr * APITCH + sc]);
  const unsigned qdst1 = lds_off(&Qs[1][sr * APITCH + sc]);
  const unsigned kdst0 = lds_off(&Ks[0][sr * APITCH + sc]);
  const unsigned kdst1 = lds_off(&Ks[1][sr * APITCH + sc]);
  const int nkb = qb + 1;   // causal: key block <= query block
  for (int kb = 0; kb < nkb; ++kb) {
    const size_t krow = (size_t)(kb * 64 + sr) * DQK + sc;
    v8f acc0 = {}, acc1 = {};
    async_ld_b128(qdst0, &qc[qrow]);
    async_ld_b128(kdst0, &kv[krow]);
    wait_async0();
    __syncthreads();
    int cur = 0;
    for (int k0 = 32; k0 <= DQK; k0 += 32) {
      const bool more = (k0 < DQK);
      if (more) {
        async_ld_b128(cur ? qdst0 : qdst1, &qc[qrow + k0]);
        async_ld_b128(cur ? kdst0 : kdst1, &kv[krow + k0]);
      }
      v16bf a  = load_frag_a(&Qs[cur][(wm * 16) * APITCH], APITCH);
      v16bf b0 = load_frag_a(&Ks[cur][(wn * 32) * APITCH], APITCH);
      v16bf b1 = load_frag_a(&Ks[cur][(wn * 32 + 16) * APITCH], APITCH);
      acc0 = WMMA_BF16(a, b0, acc0);
      acc1 = WMMA_BF16(a, b1, acc1);
      if (more) wait_async0();
      __syncthreads();
      cur ^= 1;
    }
    // scale + causal mask + row max
#pragma unroll
    for (int i = 0; i < 8; ++i) {
      int row = wm * 16 + half * 8 + i;
      int gq  = q0 + row;
      int c0  = kb * 64 + wn * 32 + cl;
      float s0 = acc0[i] * SCALE_ATTN; if (c0 > gq)      s0 = -__builtin_inff();
      float s1 = acc1[i] * SCALE_ATTN; if (c0 + 16 > gq) s1 = -__builtin_inff();
      acc0[i] = s0; acc1[i] = s1;
      float v = fmaxf(s0, s1);
      v = fmaxf(v, __shfl_xor(v, 1));
      v = fmaxf(v, __shfl_xor(v, 2));
      v = fmaxf(v, __shfl_xor(v, 4));
      v = fmaxf(v, __shfl_xor(v, 8));
      if (cl == 0) pmax[wn][row] = v;
    }
    __syncthreads();
    if (tid < 64) {
      float tm = fmaxf(pmax[0][tid], pmax[1][tid]);
      mnew[tid] = fmaxf(mrun[tid], tm);
    }
    __syncthreads();
#pragma unroll
    for (int i = 0; i < 8; ++i) {
      int row = wm * 16 + half * 8 + i;
      float nm = mnew[row];
      float v = __expf(acc0[i] - nm) + __expf(acc1[i] - nm);
      v += __shfl_xor(v, 1);
      v += __shfl_xor(v, 2);
      v += __shfl_xor(v, 4);
      v += __shfl_xor(v, 8);
      if (cl == 0) psum[wn][row] = v;
    }
    __syncthreads();
    if (tid < 64) {
      float nm = mnew[tid];
      float l = lrun[tid] * __expf(mrun[tid] - nm) + psum[0][tid] + psum[1][tid];
      lrun[tid] = l; mrun[tid] = nm;
    }
    __syncthreads();
  }
  if (tid < 64) {
    mOut[(size_t)(q0 + tid) * NHEAD + n] = mrun[tid];
    lOut[(size_t)(q0 + tid) * NHEAD + n] = lrun[tid];
  }
}

// ---------------- attention pass 2: O = softmax(S)·C_kv ----------------
__global__ __launch_bounds__(256) void attn_pass2(
    const u16* __restrict__ qc, const u16* __restrict__ kv,
    const float* __restrict__ mIn, const float* __restrict__ lIn,
    u16* __restrict__ olat) {
  const int qb = blockIdx.x, n = blockIdx.y;
  const int q0 = qb * 32;
  __shared__ __align__(16) u16 Qs[2][32 * APITCH];
  __shared__ __align__(16) u16 Ks[2][64 * APITCH];
  __shared__ __align__(16) u16 Ps[32 * 66];
  __shared__ __align__(16) u16 Bs[32 * 520];
  __shared__ float ms[32], lis[32];
  const int tid = threadIdx.x;
  const int wid = tid >> 5, lane = tid & 31;
  const int half = lane >> 4, cl = lane & 15;
  const int swm = wid >> 2, swn = wid & 3;
  const int sr = tid >> 2, sc = (tid & 3) * 8;   // staging chunk
  if (tid < 32) {
    ms[tid] = mIn[(size_t)(q0 + tid) * NHEAD + n];
    float l = lIn[(size_t)(q0 + tid) * NHEAD + n];
    lis[tid] = (l > 0.f) ? 1.f / l : 0.f;
  }
  __syncthreads();
  v8f accO[2][4];
  v8f vzero = {};
#pragma unroll
  for (int sm = 0; sm < 2; ++sm)
#pragma unroll
    for (int sn = 0; sn < 4; ++sn) accO[sm][sn] = vzero;

  const size_t qrow = ((size_t)(q0 + sr) * NHEAD + n) * DQK + sc;  // valid for tid < 128
  const unsigned qdst0 = (tid < 128) ? lds_off(&Qs[0][sr * APITCH + sc]) : 0u;
  const unsigned qdst1 = (tid < 128) ? lds_off(&Qs[1][sr * APITCH + sc]) : 0u;
  const unsigned kdst0 = lds_off(&Ks[0][sr * APITCH + sc]);
  const unsigned kdst1 = lds_off(&Ks[1][sr * APITCH + sc]);
  const int nkb = qb / 2 + 1;
  for (int kb = 0; kb < nkb; ++kb) {
    const size_t krow = (size_t)(kb * 64 + sr) * DQK + sc;
    // --- S = Q·K^T (async double-buffered over 576-deep K dim) ---
    v8f accS = {};
    if (tid < 128) async_ld_b128(qdst0, &qc[qrow]);
    async_ld_b128(kdst0, &kv[krow]);
    wait_async0();
    __syncthreads();
    int cur = 0;
    for (int k0 = 32; k0 <= DQK; k0 += 32) {
      const bool more = (k0 < DQK);
      if (more) {
        if (tid < 128) async_ld_b128(cur ? qdst0 : qdst1, &qc[qrow + k0]);
        async_ld_b128(cur ? kdst0 : kdst1, &kv[krow + k0]);
      }
      v16bf a = load_frag_a(&Qs[cur][(swm * 16) * APITCH], APITCH);
      v16bf b = load_frag_a(&Ks[cur][(swn * 16) * APITCH], APITCH);
      accS = WMMA_BF16(a, b, accS);
      if (more) wait_async0();
      __syncthreads();
      cur ^= 1;
    }
    // --- P = exp(scale*s - m)/l (causal-masked), bf16 into LDS ---
#pragma unroll
    for (int i = 0; i < 8; ++i) {
      int row = swm * 16 + half * 8 + i;
      int col = swn * 16 + cl;
      int gq = q0 + row, gs = kb * 64 + col;
      float p = 0.f;
      if (gs <= gq) p = __expf(accS[i] * SCALE_ATTN - ms[row]) * lis[row];
      Ps[row * 66 + col] = f2b(p);
    }
    __syncthreads();
    // --- O += P(32x64) · C_kv(64x512) ---
    for (int ks = 0; ks < 64; ks += 32) {
#pragma unroll
      for (int j = 0; j < 8; ++j) {
        int c = tid + j * 256;
        int r = c >> 6, off = (c & 63) * 8;
        *(uint4*)&Bs[r * 520 + off] =
            *(const uint4*)&kv[(size_t)(kb * 64 + ks + r) * DQK + off];
      }
      __syncthreads();
      v16bf ap0 = load_frag_a(&Ps[0 * 66 + ks], 66);
      v16bf ap1 = load_frag_a(&Ps[16 * 66 + ks], 66);
#pragma unroll
      for (int sn = 0; sn < 4; ++sn) {
        v16bf b = load_frag_b_rm(&Bs[wid * 64 + sn * 16], 520);
        accO[0][sn] = WMMA_BF16(ap0, b, accO[0][sn]);
        accO[1][sn] = WMMA_BF16(ap1, b, accO[1][sn]);
      }
      __syncthreads();
    }
  }
  // store o_latent bf16 [T, N, 512]
#pragma unroll
  for (int sm = 0; sm < 2; ++sm)
#pragma unroll
    for (int sn = 0; sn < 4; ++sn)
#pragma unroll
      for (int i = 0; i < 8; ++i) {
        int row = sm * 16 + half * 8 + i;
        int col = wid * 64 + sn * 16 + cl;
        olat[((size_t)(q0 + row) * NHEAD + n) * RDIM + col] = f2b(accO[sm][sn][i]);
      }
}

// ---------------- host launcher ----------------
extern "C" void kernel_launch(void* const* d_in, const int* in_sizes, int n_in,
                              void* d_out, int out_size, void* d_ws, size_t ws_size,
                              hipStream_t stream) {
  (void)in_sizes; (void)n_in; (void)out_size; (void)ws_size;
  const float* hs      = (const float*)d_in[0];
  const float* q_a_w   = (const float*)d_in[1];
  const float* q_a_ln  = (const float*)d_in[2];
  const float* q_b_w   = (const float*)d_in[3];
  const float* kv_a_w  = (const float*)d_in[4];
  const float* kv_a_ln = (const float*)d_in[5];
  const float* kv_b_w  = (const float*)d_in[6];
  const float* o_w     = (const float*)d_in[7];
  // d_in[8] = positions == arange(T), folded into the kernels.

  float* out0    = (float*)d_out;                       // [T, H]
  float* kvfused = out0 + (size_t)T_SEQ * H_DIM;        // [T, 576]

  char* p = (char*)d_ws;
  auto alloc = [&](size_t bytes) -> void* {
    void* r = (void*)p;
    p += (bytes + 255) & ~(size_t)255;
    return r;
  };
  typedef unsigned short u16h;
  u16h*  hs_bf   = (u16h*)alloc((size_t)T_SEQ * H_DIM * 2);
  u16h*  qaw_bf  = (u16h*)alloc((size_t)H_DIM * QLORA * 2);
  u16h*  qbw_bf  = (u16h*)alloc((size_t)QLORA * NHEAD * QKH * 2);
  u16h*  kvaw_bf = (u16h*)alloc((size_t)H_DIM * DQK * 2);
  u16h*  kvbw_bf = (u16h*)alloc((size_t)RDIM * NHEAD * 256 * 2);
  u16h*  ow_bf   = (u16h*)alloc((size_t)NHEAD * 128 * H_DIM * 2);
  float* qacc    = (float*)alloc((size_t)T_SEQ * QLORA * 4);
  u16h*  qc_bf   = (u16h*)alloc((size_t)T_SEQ * QLORA * 2);
  float* qfull   = (float*)alloc((size_t)T_SEQ * NHEAD * QKH * 4);
  u16h*  qnope   = (u16h*)alloc((size_t)T_SEQ * NHEAD * DKN * 2);
  u16h*  combq   = (u16h*)alloc((size_t)T_SEQ * NHEAD * DQK * 2);
  float* kva     = (float*)alloc((size_t)T_SEQ * DQK * 4);
  u16h*  kv_bf   = (u16h*)alloc((size_t)T_SEQ * DQK * 2);
  float* mbuf    = (float*)alloc((size_t)T_SEQ * NHEAD * 4);
  float* lbuf    = (float*)alloc((size_t)T_SEQ * NHEAD * 4);
  u16h*  olat    = (u16h*)alloc((size_t)T_SEQ * NHEAD * RDIM * 2);
  u16h*  ov_bf   = (u16h*)alloc((size_t)T_SEQ * NHEAD * 128 * 2);

  dim3 blk(256);
  auto cast = [&](const float* src, u16h* dst, long long nel) {
    cast_f32_bf16<<<dim3(1024), blk, 0, stream>>>(src, dst, nel / 8);
  };
  cast(hs,     hs_bf,   (long long)T_SEQ * H_DIM);
  cast(q_a_w,  qaw_bf,  (long long)H_DIM * QLORA);
  cast(q_b_w,  qbw_bf,  (long long)QLORA * NHEAD * QKH);
  cast(kv_a_w, kvaw_bf, (long long)H_DIM * DQK);
  cast(kv_b_w, kvbw_bf, (long long)RDIM * NHEAD * 256);
  cast(o_w,    ow_bf,   (long long)NHEAD * 128 * H_DIM);

  // q_a: [T,H] x [H,QLORA] -> fp32
  gemm_bf16<false, false><<<dim3(QLORA / 128, T_SEQ / 128, 1), blk, 0, stream>>>(
      hs_bf, qaw_bf, qacc, T_SEQ, QLORA, H_DIM, H_DIM, QLORA, QLORA, 0, 0, 0);
  // rmsnorm -> q_c bf16
  rmsnorm_kernel<<<dim3(T_SEQ), blk, 0, stream>>>(qacc, QLORA, QLORA, q_a_ln,
                                                  qc_bf, QLORA, (float*)nullptr, 0);
  // q_b: [T,QLORA] x [QLORA, N*192] -> fp32 qfull
  gemm_bf16<false, false><<<dim3((NHEAD * QKH) / 128, T_SEQ / 128, 1), blk, 0, stream>>>(
      qc_bf, qbw_bf, qfull, T_SEQ, NHEAD * QKH, QLORA, QLORA, NHEAD * QKH, NHEAD * QKH, 0, 0, 0);
  // split + RoPE(q_pe) -> qnope bf16, combq[.,512:576]
  q_post<<<dim3(T_SEQ, NHEAD), dim3(64), 0, stream>>>(qfull, qnope, combq);

  // kv_a: [T,H] x [H,576] -> fp32
  gemm_bf16<false, false><<<dim3((DQK + 127) / 128, T_SEQ / 128, 1), blk, 0, stream>>>(
      hs_bf, kvaw_bf, kva, T_SEQ, DQK, H_DIM, H_DIM, DQK, DQK, 0, 0, 0);
  // rmsnorm(c_kv) -> kv_bf[0:512] bf16 and kv_fused[0:512] fp32
  rmsnorm_kernel<<<dim3(T_SEQ), blk, 0, stream>>>(kva, DQK, RDIM, kv_a_ln,
                                                  kv_bf, DQK, kvfused, DQK);
  // RoPE(k_pe) -> kv_bf[512:576] bf16 and kv_fused[512:576] fp32
  kpe_rope<<<dim3(T_SEQ), dim3(32), 0, stream>>>(kva, kv_bf, kvfused);

  // ql_nope[t,n,r] = sum_d qnope[t,n,d] * w_uk[r,n,d]  (B transposed), per head
  gemm_bf16<true, true><<<dim3(RDIM / 128, T_SEQ / 128, NHEAD), blk, 0, stream>>>(
      qnope, kvbw_bf, combq, T_SEQ, RDIM, DKN,
      NHEAD * DKN, NHEAD * 256, NHEAD * DQK,
      /*aBatch*/ DKN, /*bBatch*/ 256, /*cBatch*/ DQK);

  // attention softmax stats, then O = P · C_kv
  attn_pass1<<<dim3(T_SEQ / 64, NHEAD), blk, 0, stream>>>(combq, kv_bf, mbuf, lbuf);
  attn_pass2<<<dim3(T_SEQ / 32, NHEAD), blk, 0, stream>>>(combq, kv_bf, mbuf, lbuf, olat);

  // o_v[t,n,v] = sum_r olat[t,n,r] * w_uv[r,n,v], per head
  gemm_bf16<false, true><<<dim3(1, T_SEQ / 128, NHEAD), blk, 0, stream>>>(
      olat, kvbw_bf + 128, ov_bf, T_SEQ, 128, RDIM,
      NHEAD * RDIM, NHEAD * 256, NHEAD * 128,
      /*aBatch*/ RDIM, /*bBatch*/ 256, /*cBatch*/ 128);

  // output = o_v[T, N*128] x o_w[N*128, H] -> fp32 d_out
  gemm_bf16<false, false><<<dim3(H_DIM / 128, T_SEQ / 128, 1), blk, 0, stream>>>(
      ov_bf, ow_bf, out0, T_SEQ, H_DIM, NHEAD * 128,
      NHEAD * 128, H_DIM, H_DIM, 0, 0, 0);
}
